// GraphCNN_P_25709674233955
// MI455X (gfx1250) — matrix operand order repaired and checked by reading, not compiled, not run on hardware
//
#include <hip/hip_runtime.h>
#include <math.h>

#define N_NODES 16000
#define DIM     512
#define NEDGE   256000
#define BG      16
#define NPG     1000
#define HEADS   8
#define DH      64
#define DGOUT   256
#define LN_EPS  1e-5f

typedef __attribute__((ext_vector_type(16))) _Float16 v16h;
typedef __attribute__((ext_vector_type(8)))  _Float16 h8;
typedef __attribute__((ext_vector_type(8)))  float    v8f;

// ---------------- elementwise helpers ----------------
__global__ void k_copy(const float* __restrict__ a, float* __restrict__ o, int n) {
    int i = blockIdx.x * 256 + threadIdx.x;
    if (i < n) o[i] = a[i];
}
__global__ void k_zero(float* o, int n) {
    int i = blockIdx.x * 256 + threadIdx.x;
    if (i < n) o[i] = 0.0f;
}
// f32 -> f16 bulk convert (activations before each GEMM)
__global__ void k_cvt_f16(const float* __restrict__ a, _Float16* __restrict__ o, int n) {
    int i = blockIdx.x * 256 + threadIdx.x;
    if (i < n) o[i] = (_Float16)a[i];
}
// transposed + converted weights: WT[n*K + k] = (f16) W[k*Nout + n]
__global__ void k_transpose_f16(const float* __restrict__ W, _Float16* __restrict__ WT,
                                int K, int Nout) {
    int i = blockIdx.x * 256 + threadIdx.x;
    if (i >= K * Nout) return;
    int k = i % K, n = i / K;
    WT[(size_t)n * K + k] = (_Float16)W[(size_t)k * Nout + n];
}

// ---------------- WMMA GEMM ----------------
// C[M x Nout] = [addres ? res + ] relu?( A[M x K] @ W + bias )
// A is f16 [M x K]; W is transposed+f16: WT[Nout x K].
// One wave per 16(M) x 128(N) strip: 8 accumulators reuse each A tile 8x.
// All 8 B-tiles of a k-step are loaded into distinct registers BEFORE the WMMA
// burst so load latency overlaps the matrix pipe (staggered loadcnt waits).
// REQUIRES: M % 16 == 0, K % 32 == 0, Nout % 128 == 0.
__global__ __launch_bounds__(32) void k_gemm_wmma(
    const _Float16* __restrict__ A, const _Float16* __restrict__ WT,
    const float* __restrict__ bias, const float* __restrict__ res,
    float* __restrict__ C, int M, int K, int Nout, int relu, int addres)
{
    int lane = threadIdx.x & 31;
    int half = lane >> 4;          // 0 or 1
    int l16  = lane & 15;
    int m0 = blockIdx.x * 16;
    int n0 = blockIdx.y * 128;

    v8f acc[8];
    #pragma unroll
    for (int t = 0; t < 8; ++t) acc[t] = (v8f){0.f,0.f,0.f,0.f,0.f,0.f,0.f,0.f};

    const _Float16* Ap = A + (size_t)(m0 + l16) * K;
    // per n-tile t: b[j] = WT[(n0+16t+l16)*K + k0 + 8*half + j] (j<8), k0 + 16 + 8*half + (j-8) (j>=8)
    const _Float16* Wp[8];
    #pragma unroll
    for (int t = 0; t < 8; ++t) Wp[t] = WT + (size_t)(n0 + 16 * t + l16) * K + 8 * half;

    for (int k0 = 0; k0 < K; k0 += 32) {
        // A 16x32 f16 layout: lane holds row M=l16; elements 0..7 = K 8h..8h+7,
        // elements 8..15 = K 16+8h..+7 -> two contiguous 16-byte chunks.
        h8 alo = *(const h8*)(Ap + k0 + 8 * half);
        h8 ahi = *(const h8*)(Ap + k0 + 16 + 8 * half);
        v16h a = __builtin_shufflevector(alo, ahi, 0,1,2,3,4,5,6,7,8,9,10,11,12,13,14,15);

        // stage all 8 B tiles first (16 b128 loads in flight)
        h8 blo[8], bhi[8];
        #pragma unroll
        for (int t = 0; t < 8; ++t) {
            blo[t] = *(const h8*)(Wp[t] + k0);
            bhi[t] = *(const h8*)(Wp[t] + k0 + 16);
        }
        // then burst the 8 WMMAs against them
        #pragma unroll
        for (int t = 0; t < 8; ++t) {
            v16h b = __builtin_shufflevector(blo[t], bhi[t],
                                             0,1,2,3,4,5,6,7,8,9,10,11,12,13,14,15);
            acc[t] = __builtin_amdgcn_wmma_f32_16x16x32_f16(
                false, a, false, b, (short)0, acc[t], false, false);
        }
    }

    // C/D layout: VGPR r holds M = r + 8*half, N = l16 across lanes
    #pragma unroll
    for (int t = 0; t < 8; ++t) {
        int n = n0 + 16 * t + l16;
        float bv = bias ? bias[n] : 0.0f;
        #pragma unroll
        for (int r = 0; r < 8; ++r) {
            int m = m0 + r + 8 * half;
            float v = acc[t][r] + bv;
            if (relu) v = fmaxf(v, 0.0f);
            if (addres) v += res[(size_t)m * Nout + n];
            C[(size_t)m * Nout + n] = v;
        }
    }
}

// ---------------- graph ops ----------------
// agg[dst] += h[src]  (one block per edge, 256 threads x 2 dims)
__global__ void k_scatter_add(const float* __restrict__ h, const int* __restrict__ src,
                              const int* __restrict__ dst, float* agg)
{
    int e = blockIdx.x, d = threadIdx.x;
    int s = src[e], t = dst[e];
    atomicAdd(&agg[(size_t)t * DIM + d],       h[(size_t)s * DIM + d]);
    atomicAdd(&agg[(size_t)t * DIM + d + 256], h[(size_t)s * DIM + d + 256]);
}
__global__ void k_degree(const int* __restrict__ dst, float* deg) {
    int e = blockIdx.x * 256 + threadIdx.x;
    if (e < NEDGE) atomicAdd(&deg[dst[e]], 1.0f);
}
__global__ void k_rsqrt_deg(float* d) {  // in place: dis = rsqrt(deg + 1)
    int i = blockIdx.x * 256 + threadIdx.x;
    if (i < N_NODES) d[i] = rsqrtf(d[i] + 1.0f);
}
// out = xw * dis^2 + bias   (self-loop term + bias, initializes out)
__global__ void k_gcn_self(const float* __restrict__ xw, const float* __restrict__ dis,
                           const float* __restrict__ b, float* out)
{
    int idx = blockIdx.x * 256 + threadIdx.x;
    if (idx >= N_NODES * DIM) return;
    int i = idx / DIM, d = idx % DIM;
    float s = dis[i];
    out[idx] = xw[idx] * s * s + b[d];
}
// out[dst] += xw[src] * dis[src]*dis[dst]
__global__ void k_gcn_scatter(const float* __restrict__ xw, const float* __restrict__ dis,
                              const int* __restrict__ src, const int* __restrict__ dst, float* out)
{
    int e = blockIdx.x, d = threadIdx.x;
    int s = src[e], t = dst[e];
    float w = dis[s] * dis[t];
    atomicAdd(&out[(size_t)t * DIM + d],       xw[(size_t)s * DIM + d] * w);
    atomicAdd(&out[(size_t)t * DIM + d + 256], xw[(size_t)s * DIM + d + 256] * w);
}

// ---------------- normalization ----------------
__global__ void k_bn_stats(const float* __restrict__ x, float* mean, float* var) {
    int d = blockIdx.x, tid = threadIdx.x;
    __shared__ float s1[256], s2[256];
    float a = 0.f, b = 0.f;
    for (int i = tid; i < N_NODES; i += 256) {
        float v = x[(size_t)i * DIM + d];
        a += v; b += v * v;
    }
    s1[tid] = a; s2[tid] = b; __syncthreads();
    for (int o = 128; o > 0; o >>= 1) {
        if (tid < o) { s1[tid] += s1[tid + o]; s2[tid] += s2[tid + o]; }
        __syncthreads();
    }
    if (tid == 0) {
        float m = s1[0] / (float)N_NODES;
        mean[d] = m;
        var[d]  = s2[0] / (float)N_NODES - m * m;
    }
}
__global__ void k_bn_apply(const float* __restrict__ x, const float* __restrict__ mean,
                           const float* __restrict__ var, const float* __restrict__ g,
                           const float* __restrict__ b, float* out)
{
    int idx = blockIdx.x * 256 + threadIdx.x;
    if (idx >= N_NODES * DIM) return;
    int d = idx % DIM;
    out[idx] = (x[idx] - mean[d]) * rsqrtf(var[d] + LN_EPS) * g[d] + b[d];
}
__global__ void k_layernorm(const float* __restrict__ in, const float* __restrict__ g,
                            const float* __restrict__ b, float* out)
{
    int r = blockIdx.x, tid = threadIdx.x;
    __shared__ float s1[256], s2[256];
    float a = 0.f, q = 0.f;
    for (int d = tid; d < DIM; d += 256) {
        float v = in[(size_t)r * DIM + d];
        a += v; q += v * v;
    }
    s1[tid] = a; s2[tid] = q; __syncthreads();
    for (int o = 128; o > 0; o >>= 1) {
        if (tid < o) { s1[tid] += s1[tid + o]; s2[tid] += s2[tid + o]; }
        __syncthreads();
    }
    float m  = s1[0] / (float)DIM;
    float vr = s2[0] / (float)DIM - m * m;
    float is = rsqrtf(vr + LN_EPS);
    for (int d = tid; d < DIM; d += 256)
        out[(size_t)r * DIM + d] = (in[(size_t)r * DIM + d] - m) * is * g[d] + b[d];
}

// ---------------- small dense ops ----------------
// Q[d] = q_b[d] + sum_k S2[k] * q_w[k*DIM + d]   (single 1x512 projection)
__global__ void k_qproj(const float* __restrict__ S2, const float* __restrict__ qw,
                        const float* __restrict__ qb, float* Q)
{
    int d = blockIdx.x * 256 + threadIdx.x;
    if (d >= DIM) return;
    float acc = qb[d];
    for (int k = 0; k < DIM; ++k) acc += S2[k] * qw[(size_t)k * DIM + d];
    Q[d] = acc;
}
// single-seed attention per (graph, head): out[b, h*64+d] = q + softmax(qK^T/sqrt(D)) V
__global__ void k_attn(const float* __restrict__ Q, const float* __restrict__ Kb,
                       const float* __restrict__ Vb, float* out)
{
    int bh = blockIdx.x;
    int b = bh / HEADS, h = bh % HEADS;
    int tid = threadIdx.x;
    __shared__ float sc[NPG];
    __shared__ float red[256];
    __shared__ float qs[DH];
    if (tid < DH) qs[tid] = Q[h * DH + tid];
    __syncthreads();
    const float scale = 0.04419417382415922f;  // 1/sqrt(512)
    float lmax = -1e30f;
    for (int k = tid; k < NPG; k += 256) {
        const float* kp = Kb + ((size_t)(b * NPG + k)) * DIM + h * DH;
        float dacc = 0.f;
        for (int d = 0; d < DH; ++d) dacc += qs[d] * kp[d];
        dacc *= scale;
        sc[k] = dacc;
        lmax = fmaxf(lmax, dacc);
    }
    red[tid] = lmax; __syncthreads();
    for (int o = 128; o > 0; o >>= 1) {
        if (tid < o) red[tid] = fmaxf(red[tid], red[tid + o]);
        __syncthreads();
    }
    float mx = red[0]; __syncthreads();
    float lsum = 0.f;
    for (int k = tid; k < NPG; k += 256) {
        float p = __expf(sc[k] - mx);
        sc[k] = p; lsum += p;
    }
    red[tid] = lsum; __syncthreads();
    for (int o = 128; o > 0; o >>= 1) {
        if (tid < o) red[tid] += red[tid + o];
        __syncthreads();
    }
    float inv = 1.0f / red[0];
    __syncthreads();
    if (tid < DH) {
        float acc = 0.f;
        for (int k = 0; k < NPG; ++k)
            acc += sc[k] * Vb[((size_t)(b * NPG + k)) * DIM + h * DH + tid];
        out[(size_t)b * DIM + h * DH + tid] = qs[tid] + acc * inv;
    }
}

// ---------------- host ----------------
extern "C" void kernel_launch(void* const* d_in, const int* in_sizes, int n_in,
                              void* d_out, int out_size, void* d_ws, size_t ws_size,
                              hipStream_t stream)
{
    const float* x      = (const float*)d_in[0];
    const int*   ei     = (const int*)d_in[1];
    const int*   src    = ei;
    const int*   dst    = ei + NEDGE;
    const float* gin_w1 = (const float*)d_in[3];
    const float* gin_b1 = (const float*)d_in[4];
    const float* gin_w2 = (const float*)d_in[5];
    const float* gin_b2 = (const float*)d_in[6];
    const float* bn_g   = (const float*)d_in[7];
    const float* bn_b   = (const float*)d_in[8];
    const float* lin1_w = (const float*)d_in[9];
    const float* lin1_b = (const float*)d_in[10];
    const float* lin2_w = (const float*)d_in[11];
    const float* lin2_b = (const float*)d_in[12];
    const float* S2     = (const float*)d_in[14];
    const float* q_w    = (const float*)d_in[15];
    const float* q_b    = (const float*)d_in[16];
    const float* k_w    = (const float*)d_in[17];
    const float* k_b    = (const float*)d_in[18];
    const float* v_w    = (const float*)d_in[19];
    const float* v_b    = (const float*)d_in[20];
    const float* o_w    = (const float*)d_in[21];
    const float* o_b    = (const float*)d_in[22];
    const float* ln0_g  = (const float*)d_in[23];
    const float* ln0_b  = (const float*)d_in[24];
    const float* ln1_g  = (const float*)d_in[25];
    const float* ln1_b  = (const float*)d_in[26];
    float* out = (float*)d_out;

    const size_t ND = (size_t)N_NODES * DIM;
    const size_t DD = (size_t)DIM * DIM;
    float* ws   = (float*)d_ws;
    float* hbuf = ws;              // [N, D]
    float* agg  = ws + ND;         // [N, D]
    float* t1   = ws + 2 * ND;     // [N, D]  (also xl)
    float* t2   = ws + 3 * ND;     // [N, D]  (also V)
    float* xtra = ws + 4 * ND;
    float* Qbuf = xtra;            // [D]
    float* o1   = Qbuf + DIM;      // [B, D]
    float* o2   = o1 + BG * DIM;   // [B, D]
    float* bx   = o2 + BG * DIM;   // [B, D]
    float* dis  = bx + BG * DIM;   // [N]
    float* mean = dis + N_NODES;   // [D]
    float* var  = mean + DIM;      // [D]
    _Float16* wt_h = (_Float16*)(var + DIM);        // [D, D] f16 transposed weight
    _Float16* ah   = (_Float16*)(var + DIM + DD/2 + 64);  // [N, D] f16 activations

    const int ndBlocks = (int)((ND + 255) / 256);
    const dim3 gBig(N_NODES / 16, DIM / 128);      // 1000 x 4
    const dim3 gSmall(BG / 16, DIM / 128);         // 1 x 4
    const dim3 wBlk(32);
    const int tBlocks = (int)((DD + 255) / 256);
    const int sBlocks = (int)(((size_t)BG * DIM + 255) / 256);

    // h = x
    k_copy<<<ndBlocks, 256, 0, stream>>>(x, hbuf, (int)ND);

    // ---- 3 GIN layers ----
    for (int i = 0; i < 3; ++i) {
        k_copy<<<ndBlocks, 256, 0, stream>>>(hbuf, agg, (int)ND);
        k_scatter_add<<<NEDGE, 256, 0, stream>>>(hbuf, src, dst, agg);
        // t1 = relu(agg @ w1 + b1)
        k_cvt_f16<<<ndBlocks, 256, 0, stream>>>(agg, ah, (int)ND);
        k_transpose_f16<<<tBlocks, 256, 0, stream>>>(gin_w1 + (size_t)i * DD, wt_h, DIM, DIM);
        k_gemm_wmma<<<gBig, wBlk, 0, stream>>>(ah, wt_h, gin_b1 + (size_t)i * DIM, nullptr,
                                               t1, N_NODES, DIM, DIM, 1, 0);
        // t2 = relu(t1 @ w2 + b2) [+ h residual for i>0]
        k_cvt_f16<<<ndBlocks, 256, 0, stream>>>(t1, ah, (int)ND);
        k_transpose_f16<<<tBlocks, 256, 0, stream>>>(gin_w2 + (size_t)i * DD, wt_h, DIM, DIM);
        k_gemm_wmma<<<gBig, wBlk, 0, stream>>>(ah, wt_h, gin_b2 + (size_t)i * DIM, hbuf,
                                               t2, N_NODES, DIM, DIM, 1, (i > 0) ? 1 : 0);
        // h = BN(t2)
        k_bn_stats<<<DIM, 256, 0, stream>>>(t2, mean, var);
        k_bn_apply<<<ndBlocks, 256, 0, stream>>>(t2, mean, var,
                                                 bn_g + (size_t)i * DIM, bn_b + (size_t)i * DIM, hbuf);
    }
    // n_level output
    k_copy<<<ndBlocks, 256, 0, stream>>>(hbuf, out, (int)ND);

    // ---- xl = h @ lin1 + b ----  (h f16 copy reused for lin1 GEMM)
    k_cvt_f16<<<ndBlocks, 256, 0, stream>>>(hbuf, ah, (int)ND);
    k_transpose_f16<<<tBlocks, 256, 0, stream>>>(lin1_w, wt_h, DIM, DIM);
    k_gemm_wmma<<<gBig, wBlk, 0, stream>>>(ah, wt_h, lin1_b, nullptr, t1, N_NODES, DIM, DIM, 0, 0);

    // ---- symmetric GCN normalization factors ----
    k_zero<<<(N_NODES + 255) / 256, 256, 0, stream>>>(dis, N_NODES);
    k_degree<<<(NEDGE + 255) / 256, 256, 0, stream>>>(dst, dis);
    k_rsqrt_deg<<<(N_NODES + 255) / 256, 256, 0, stream>>>(dis);

    // (pool 0 is dead code in the reference: only p=1 reaches the output)
    // xl f16 copy feeds both K and V projections
    k_cvt_f16<<<ndBlocks, 256, 0, stream>>>(t1, ah, (int)ND);
    // K = GCN(xl; k_w[1], k_b[1])  -> hbuf
    k_transpose_f16<<<tBlocks, 256, 0, stream>>>(k_w + DD, wt_h, DIM, DIM);
    k_gemm_wmma<<<gBig, wBlk, 0, stream>>>(ah, wt_h, nullptr, nullptr, agg, N_NODES, DIM, DIM, 0, 0);
    k_gcn_self<<<ndBlocks, 256, 0, stream>>>(agg, dis, k_b + DIM, hbuf);
    k_gcn_scatter<<<NEDGE, 256, 0, stream>>>(agg, dis, src, dst, hbuf);
    // V = GCN(xl; v_w[1], v_b[1])  -> t2
    k_transpose_f16<<<tBlocks, 256, 0, stream>>>(v_w + DD, wt_h, DIM, DIM);
    k_gemm_wmma<<<gBig, wBlk, 0, stream>>>(ah, wt_h, nullptr, nullptr, agg, N_NODES, DIM, DIM, 0, 0);
    k_gcn_self<<<ndBlocks, 256, 0, stream>>>(agg, dis, v_b + DIM, t2);
    k_gcn_scatter<<<NEDGE, 256, 0, stream>>>(agg, dis, src, dst, t2);

    // Q = S2 @ q_w[1] + q_b[1]
    k_qproj<<<(DIM + 255) / 256, 256, 0, stream>>>(S2, q_w + DD, q_b + DIM, Qbuf);

    // attention: o1[b, :] = q + softmax(qK^T) V, per head
    k_attn<<<BG * HEADS, 256, 0, stream>>>(Qbuf, hbuf, t2, o1);

    // ln0 -> o2
    k_layernorm<<<BG, 256, 0, stream>>>(o1, ln0_g + DIM, ln0_b + DIM, o2);
    // o1 = o2 + relu(o2 @ o_w[1] + o_b[1])
    k_cvt_f16<<<sBlocks, 256, 0, stream>>>(o2, ah, BG * DIM);
    k_transpose_f16<<<tBlocks, 256, 0, stream>>>(o_w + DD, wt_h, DIM, DIM);
    k_gemm_wmma<<<gSmall, wBlk, 0, stream>>>(ah, wt_h, o_b + DIM, o2, o1, BG, DIM, DIM, 1, 1);
    // ln1 -> bx
    k_layernorm<<<BG, 256, 0, stream>>>(o1, ln1_g + DIM, ln1_b + DIM, bx);

    // g_level = bx @ lin2 + b -> out[N*D ...]
    k_cvt_f16<<<sBlocks, 256, 0, stream>>>(bx, ah, BG * DIM);
    k_transpose_f16<<<(int)(((size_t)DIM * DGOUT + 255) / 256), 256, 0, stream>>>(lin2_w, wt_h, DIM, DGOUT);
    k_gemm_wmma<<<dim3(BG / 16, DGOUT / 128), wBlk, 0, stream>>>(
        ah, wt_h, lin2_b, nullptr, out + ND, BG, DIM, DGOUT, 0, 0);
}
